// CoupledHMNF_4063039062729
// MI455X (gfx1250) — compile-verified, matrix operand order checked by reading, not run on hardware
//
#include <hip/hip_runtime.h>
#include <hip/hip_bf16.h>
#include <math.h>

// ---------------------------------------------------------------- constants
#define B_SZ     4
#define L_SEQ    1024
#define D_MODEL  256
#define D_INNER  512
#define NHEADS   16
#define HEADDIM  32
#define D_STATE  32
#define D_CONV   4
#define T_TOK    (B_SZ * L_SEQ)          // 4096 tokens
#define THREADS  256                     // 8 wave32 per block

typedef __attribute__((ext_vector_type(16))) _Float16 v16h;
typedef __attribute__((ext_vector_type(8)))  float    v8f;

#if defined(__has_builtin)
#if __has_builtin(__builtin_amdgcn_global_load_async_to_lds_b32)
#define HAVE_ASYNC_LDS 1
#endif
#endif

typedef __attribute__((address_space(1))) int glb_int_t;   // global int
typedef __attribute__((address_space(3))) int lds_int_t;   // LDS int

__device__ __forceinline__ void wait_async0() {
#if defined(__has_builtin) && __has_builtin(__builtin_amdgcn_s_wait_asynccnt)
    __builtin_amdgcn_s_wait_asynccnt(0);
#else
    asm volatile("s_wait_asynccnt 0x0" ::: "memory");
#endif
}

__device__ __forceinline__ float silu_f(float x) {
    return x * (1.0f / (1.0f + __expf(-x)));
}
__device__ __forceinline__ float softplus_f(float x) {
    return (x > 20.0f) ? x : log1pf(__expf(x));
}

// K offset inside a 16x32 fp16 fragment for VGPR pair v (ISA 7.12.2 table)
__device__ __forceinline__ int frag_koff(int v, int hf) {
    return (v < 4) ? (hf * 8 + 2 * v) : (16 + hf * 8 + 2 * (v - 4));
}

// A fragment: row-major source, row fixed per lane, contiguous K pairs
__device__ __forceinline__ v16h load_frag_row(const float* rowPtr, int k0, int hf) {
    v16h f;
#pragma unroll
    for (int v = 0; v < 8; ++v) {
        int kb = k0 + frag_koff(v, hf);
        float2 p = *(const float2*)(rowPtr + kb);
        f[2 * v]     = (_Float16)p.x;
        f[2 * v + 1] = (_Float16)p.y;
    }
    return f;
}

// B fragment from a [64 x 16] f32 weight tile staged in LDS
__device__ __forceinline__ v16h load_frag_lds(const float* lds, int kbase,
                                              int idx, int hf) {
    v16h f;
#pragma unroll
    for (int v = 0; v < 8; ++v) {
        int kb = kbase + frag_koff(v, hf);
        f[2 * v]     = (_Float16)lds[kb * 16 + idx];
        f[2 * v + 1] = (_Float16)lds[(kb + 1) * 16 + idx];
    }
    return f;
}

// B fragment straight from global (fusion kernel): W is [K,N] row-major
__device__ __forceinline__ v16h load_frag_col(const float* W, int ldw, int col,
                                              int k0, int hf) {
    v16h f;
#pragma unroll
    for (int v = 0; v < 8; ++v) {
        int kb = k0 + frag_koff(v, hf);
        f[2 * v]     = (_Float16)W[(size_t)kb * ldw + col];
        f[2 * v + 1] = (_Float16)W[(size_t)(kb + 1) * ldw + col];
    }
    return f;
}

// ---------------------------------------------------------------- generic WMMA GEMM
// Block = 8 waves covering a 128-row M-super-tile at one 16-col N-tile.
// Weight chunk [64x16] staged in LDS via async-to-LDS, shared by all 8 waves.
// out[M,N] = op(A[M,K] @ W[K,N] + bias)   (A optionally row-flipped per batch)
// mode 0: store   mode 1: out += silu(r)   mode 2: out = r + resid
__global__ void wmma_gemm(const float* __restrict__ A, int lda, int flipA,
                          const float* __restrict__ W,
                          const float* __restrict__ bias,
                          float* __restrict__ out, int ldo,
                          const float* __restrict__ resid, int ldr,
                          int M, int N, int K, int mode) {
    __shared__ float ldsW[64 * 16];
    int lane = threadIdx.x & 31;
    int wv   = threadIdx.x >> 5;          // wave in block, 0..7
    int nTiles  = N >> 4;
    int mSupers = M >> 7;                  // 128 rows per block
    if ((int)blockIdx.x >= mSupers * nTiles) return;
    int mSuper = blockIdx.x / nTiles;
    int nTile  = blockIdx.x % nTiles;
    int mTile  = mSuper * 8 + wv;

    int hf  = lane >> 4;   // which K-half of the fragment this lane holds
    int idx = lane & 15;   // M for A-fragment, N for B-fragment and C/D

    int rowA = mTile * 16 + idx;
    if (flipA) {
        int b = rowA / L_SEQ, l = rowA % L_SEQ;
        rowA = b * L_SEQ + (L_SEQ - 1 - l);
    }
    const float* aRow = A + (size_t)rowA * lda;
    int colBase = nTile * 16;

    v8f c = {};
    for (int k0 = 0; k0 < K; k0 += 64) {
        // cooperative stage of W[k0..k0+63, colBase..colBase+15] -> LDS
#pragma unroll
        for (int e = threadIdx.x; e < 64 * 16; e += THREADS) {
            int r = e >> 4, ci = e & 15;
            const float* g = W + (size_t)(k0 + r) * N + colBase + ci;
#if defined(HAVE_ASYNC_LDS)
            __builtin_amdgcn_global_load_async_to_lds_b32(
                (glb_int_t*)(void*)g, (lds_int_t*)&ldsW[e], 0, 0);
#else
            ldsW[e] = *g;
#endif
        }
#if defined(HAVE_ASYNC_LDS)
        wait_async0();
#endif
        __syncthreads();

        if (k0 + 128 < K) __builtin_prefetch(aRow + k0 + 128, 0, 1);
        v16h af0 = load_frag_row(aRow, k0, hf);
        v16h bf0 = load_frag_lds(ldsW, 0, idx, hf);
        c = __builtin_amdgcn_wmma_f32_16x16x32_f16(false, af0, false, bf0,
                                                   (short)0, c, false, false);
        v16h af1 = load_frag_row(aRow, k0 + 32, hf);
        v16h bf1 = load_frag_lds(ldsW, 32, idx, hf);
        c = __builtin_amdgcn_wmma_f32_16x16x32_f16(false, af1, false, bf1,
                                                   (short)0, c, false, false);
        __syncthreads();
    }

#pragma unroll
    for (int r = 0; r < 8; ++r) {
        int m = mTile * 16 + hf * 8 + r;   // C/D layout: lanes 16-31 hold M+8
        int n = colBase + idx;
        float val = c[r];
        if (bias) val += bias[n];
        float* op = out + (size_t)m * ldo + n;
        if (mode == 1)      *op += silu_f(val);
        else if (mode == 2) *op  = val + resid[(size_t)m * ldr + n];
        else                *op  = val;
    }
}

// ---------------------------------------------------------------- fusion GEMM
// ctx[M,256] = [xa|xv|xl] @ fusion_W[768,256] + fusion_b   (wave per 16x16 tile)
__global__ void wmma_gemm_fusion(const float* __restrict__ xa,
                                 const float* __restrict__ xv,
                                 const float* __restrict__ xl,
                                 const float* __restrict__ W,
                                 const float* __restrict__ bias,
                                 float* __restrict__ out) {
    const int N = D_MODEL, K = 3 * D_MODEL;
    int lane = threadIdx.x & 31;
    int wave = blockIdx.x * (blockDim.x >> 5) + (threadIdx.x >> 5);
    int nTiles = N >> 4;
    if (wave >= (T_TOK >> 4) * nTiles) return;
    int mTile = wave / nTiles, nTile = wave % nTiles;
    int hf = lane >> 4, idx = lane & 15;
    int row = mTile * 16 + idx;
    int col = nTile * 16 + idx;

    v8f c = {};
    for (int k0 = 0; k0 < K; k0 += 64) {
#pragma unroll
        for (int kk = 0; kk < 2; ++kk) {
            int kbase = k0 + kk * 32;
            v16h af;
#pragma unroll
            for (int v = 0; v < 8; ++v) {
                int kb = kbase + frag_koff(v, hf);  // even; pair stays in one chunk
                const float* src = (kb < 256) ? (xa + (size_t)row * 256 + kb)
                                 : (kb < 512) ? (xv + (size_t)row * 256 + kb - 256)
                                              : (xl + (size_t)row * 256 + kb - 512);
                float2 p = *(const float2*)src;
                af[2 * v] = (_Float16)p.x;
                af[2 * v + 1] = (_Float16)p.y;
            }
            v16h bf = load_frag_col(W, N, col, kbase, hf);
            c = __builtin_amdgcn_wmma_f32_16x16x32_f16(false, af, false, bf,
                                                       (short)0, c, false, false);
        }
    }
#pragma unroll
    for (int r = 0; r < 8; ++r) {
        int m = mTile * 16 + hf * 8 + r;
        int n = nTile * 16 + idx;
        out[(size_t)m * N + n] = c[r] + bias[n];
    }
}

// ---------------------------------------------------------------- LayerNorm (wave per token)
__global__ void layernorm_k(const float* __restrict__ x,
                            const float* __restrict__ g,
                            const float* __restrict__ b,
                            float* __restrict__ u) {
    int wave = (blockIdx.x * blockDim.x + threadIdx.x) >> 5;
    int lane = threadIdx.x & 31;
    if (wave >= T_TOK) return;
    const float* xp = x + (size_t)wave * D_MODEL;
    float v[8], s = 0.f;
#pragma unroll
    for (int j = 0; j < 8; ++j) { v[j] = xp[lane + 32 * j]; s += v[j]; }
#pragma unroll
    for (int o = 16; o > 0; o >>= 1) s += __shfl_xor(s, o, 32);
    float mean = s * (1.0f / D_MODEL);
    float sq = 0.f;
#pragma unroll
    for (int j = 0; j < 8; ++j) { float d = v[j] - mean; sq += d * d; }
#pragma unroll
    for (int o = 16; o > 0; o >>= 1) sq += __shfl_xor(sq, o, 32);
    float rstd = rsqrtf(sq * (1.0f / D_MODEL) + 1e-5f);
    float* up = u + (size_t)wave * D_MODEL;
#pragma unroll
    for (int j = 0; j < 8; ++j) {
        int cc = lane + 32 * j;
        up[cc] = (v[j] - mean) * rstd * g[cc] + b[cc];
    }
}

// ---------------------------------------------------------------- causal conv + SiLU
// xb lives in zx[..., 512:1024]; dir=1 reads the time-flipped sequence
__global__ void conv_silu_k(const float* __restrict__ zx,
                            const float* __restrict__ conv_w,
                            const float* __restrict__ conv_b,
                            float* __restrict__ xcf,
                            float* __restrict__ xcb) {
    int dir = blockIdx.y;
    int i = blockIdx.x * blockDim.x + threadIdx.x;
    if (i >= T_TOK * D_INNER) return;
    int c = i & (D_INNER - 1);
    int tok = i >> 9;
    int b = tok / L_SEQ, l = tok % L_SEQ;
    float acc = conv_b[c];
#pragma unroll
    for (int k = 0; k < D_CONV; ++k) {
        int ls = l - (D_CONV - 1) + k;
        if (ls < 0) continue;
        int src = dir ? (L_SEQ - 1 - ls) : ls;
        acc += zx[((size_t)(b * L_SEQ + src)) * (2 * D_INNER) + D_INNER + c] *
               conv_w[k * D_INNER + c];
    }
    float* dst = dir ? xcb : xcf;
    dst[(size_t)tok * D_INNER + c] = silu_f(acc);
}

// ---------------------------------------------------------------- SSM scan
// one wave32 per (batch, head); lane p holds state row S[p, 0..31]
__global__ void ssm_scan_k(const float* __restrict__ xc,    // [B,L,512]
                           const float* __restrict__ dtBC,  // [B,L,80]
                           const float* __restrict__ A_log,
                           const float* __restrict__ dt_bias,
                           const float* __restrict__ Dskip,
                           float* __restrict__ y,           // [B,L,512]
                           int dir) {
    int wave = (blockIdx.x * blockDim.x + threadIdx.x) >> 5;
    int lane = threadIdx.x & 31;
    if (wave >= B_SZ * NHEADS) return;
    int b = wave >> 4;
    int h = wave & (NHEADS - 1);
    float A   = -__expf(A_log[h]);
    float bia = dt_bias[h];
    float Dh  = Dskip[h];
    float S[D_STATE];
#pragma unroll
    for (int n = 0; n < D_STATE; ++n) S[n] = 0.f;

    for (int l = 0; l < L_SEQ; ++l) {
        size_t tok = (size_t)b * L_SEQ + l;
        const float* dp = dtBC + tok * (NHEADS + 2 * D_STATE);
        float dt = softplus_f(dp[h] + bia);
        float Bv = dp[NHEADS + lane];
        float Cv = dp[NHEADS + D_STATE + lane];
        float xp = xc[tok * D_INNER + h * HEADDIM + lane];  // p = lane
        float decay = __expf(dt * A);
        float coef  = dt * xp;
        float acc = 0.f;
#pragma unroll
        for (int n = 0; n < D_STATE; ++n) {
            float Bn = __shfl(Bv, n, 32);
            float Cn = __shfl(Cv, n, 32);
            S[n] = S[n] * decay + coef * Bn;
            acc += S[n] * Cn;
        }
        float yv = acc + Dh * xp;
        size_t otok = dir ? ((size_t)b * L_SEQ + (L_SEQ - 1 - l)) : tok;
        float* op = y + otok * D_INNER + h * HEADDIM + lane;
        if (dir) *op += yv; else *op = yv;
    }
}

// ---------------------------------------------------------------- gate multiply
__global__ void gate_mul_k(const float* __restrict__ y,
                           const float* __restrict__ zx,   // z in [..., :512]
                           float* __restrict__ g) {
    int i = blockIdx.x * blockDim.x + threadIdx.x;
    if (i >= T_TOK * D_INNER) return;
    int c = i & (D_INNER - 1);
    int tok = i >> 9;
    float z = zx[(size_t)tok * (2 * D_INNER) + c];
    g[i] = y[i] * silu_f(z);
}

// ---------------------------------------------------------------- host side
struct BlockP {
    const float *A_log, *Dskip, *W_dtBC, *W_in, *W_out,
                *conv_b, *conv_w, *ctx_W, *ctx_b, *dt_bias, *ln_b, *ln_g;
};

static BlockP mk_block(void* const* d_in, int base) {
    BlockP p;
    p.A_log   = (const float*)d_in[base + 0];
    p.Dskip   = (const float*)d_in[base + 1];
    p.W_dtBC  = (const float*)d_in[base + 2];
    p.W_in    = (const float*)d_in[base + 3];
    p.W_out   = (const float*)d_in[base + 4];
    p.conv_b  = (const float*)d_in[base + 5];
    p.conv_w  = (const float*)d_in[base + 6];
    p.ctx_W   = (const float*)d_in[base + 7];
    p.ctx_b   = (const float*)d_in[base + 8];
    p.dt_bias = (const float*)d_in[base + 9];
    p.ln_b    = (const float*)d_in[base + 10];
    p.ln_g    = (const float*)d_in[base + 11];
    return p;
}

static void run_block(const BlockP& p, const float* x, const float* ctx,
                      float* out, float* u, float* zx, float* xcf, float* xcb,
                      float* dtf, float* dtb, float* ybuf, float* gbuf,
                      hipStream_t stream) {
    const int ew = T_TOK * D_INNER;        // 2M elementwise items
    const int MS = T_TOK / 128;            // 32 M-super-tiles per GEMM

    layernorm_k<<<(T_TOK * 32 + THREADS - 1) / THREADS, THREADS, 0, stream>>>(
        x, p.ln_g, p.ln_b, u);

    // zx = u @ W_in          [4096,256]x[256,1024]
    wmma_gemm<<<MS * (2 * D_INNER / 16), THREADS, 0, stream>>>(
        u, D_MODEL, 0, p.W_in, nullptr, zx, 2 * D_INNER, nullptr, 0,
        T_TOK, 2 * D_INNER, D_MODEL, 0);

    // xc = silu(causal_conv(xb)) forward & backward
    dim3 cg((ew + THREADS - 1) / THREADS, 2);
    conv_silu_k<<<cg, THREADS, 0, stream>>>(zx, p.conv_w, p.conv_b, xcf, xcb);

    // xc += silu(ctx @ ctx_W + ctx_b)   (backward uses flipped ctx rows)
    wmma_gemm<<<MS * (D_INNER / 16), THREADS, 0, stream>>>(
        ctx, D_MODEL, 0, p.ctx_W, p.ctx_b, xcf, D_INNER, nullptr, 0,
        T_TOK, D_INNER, D_MODEL, 1);
    wmma_gemm<<<MS * (D_INNER / 16), THREADS, 0, stream>>>(
        ctx, D_MODEL, 1, p.ctx_W, p.ctx_b, xcb, D_INNER, nullptr, 0,
        T_TOK, D_INNER, D_MODEL, 1);

    // dtBC = xc @ W_dtBC     [4096,512]x[512,80]  (80 = 5 tiles of 16)
    wmma_gemm<<<MS * 5, THREADS, 0, stream>>>(
        xcf, D_INNER, 0, p.W_dtBC, nullptr, dtf, 80, nullptr, 0,
        T_TOK, 80, D_INNER, 0);
    wmma_gemm<<<MS * 5, THREADS, 0, stream>>>(
        xcb, D_INNER, 0, p.W_dtBC, nullptr, dtb, 80, nullptr, 0,
        T_TOK, 80, D_INNER, 0);

    // sequential scans: forward writes, backward flips + accumulates
    ssm_scan_k<<<(B_SZ * NHEADS * 32 + THREADS - 1) / THREADS, THREADS, 0, stream>>>(
        xcf, dtf, p.A_log, p.dt_bias, p.Dskip, ybuf, 0);
    ssm_scan_k<<<(B_SZ * NHEADS * 32 + THREADS - 1) / THREADS, THREADS, 0, stream>>>(
        xcb, dtb, p.A_log, p.dt_bias, p.Dskip, ybuf, 1);

    // g = y * silu(z)
    gate_mul_k<<<(ew + THREADS - 1) / THREADS, THREADS, 0, stream>>>(ybuf, zx, gbuf);

    // out = g @ W_out + x
    wmma_gemm<<<MS * (D_MODEL / 16), THREADS, 0, stream>>>(
        gbuf, D_INNER, 0, p.W_out, nullptr, out, D_MODEL, x, D_MODEL,
        T_TOK, D_MODEL, D_INNER, 2);
}

extern "C" void kernel_launch(void* const* d_in, const int* in_sizes, int n_in,
                              void* d_out, int out_size, void* d_ws, size_t ws_size,
                              hipStream_t stream) {
    (void)in_sizes; (void)n_in; (void)out_size; (void)ws_size;
    const float* xa = (const float*)d_in[0];
    const float* xv = (const float*)d_in[1];
    const float* xl = (const float*)d_in[2];
    // JAX tree leaves, dict keys sorted: layer{'a',fusion_W,fusion_b,'l','v'}
    BlockP pa = mk_block(d_in, 3);
    const float* fusion_W = (const float*)d_in[15];
    const float* fusion_b = (const float*)d_in[16];
    BlockP pl = mk_block(d_in, 17);
    BlockP pv = mk_block(d_in, 29);

    float* ws   = (float*)d_ws;
    float* ctx  = ws;                          // T*256
    float* u    = ctx  + (size_t)T_TOK * 256;  // T*256
    float* zx   = u    + (size_t)T_TOK * 256;  // T*1024
    float* xcf  = zx   + (size_t)T_TOK * 1024; // T*512
    float* xcb  = xcf  + (size_t)T_TOK * 512;  // T*512
    float* dtf  = xcb  + (size_t)T_TOK * 512;  // T*80
    float* dtb  = dtf  + (size_t)T_TOK * 80;   // T*80
    float* ybuf = dtb  + (size_t)T_TOK * 80;   // T*512
    float* gbuf = ybuf + (size_t)T_TOK * 512;  // T*512

    float* out_a = (float*)d_out;
    float* out_v = out_a + (size_t)T_TOK * D_MODEL;
    float* out_l = out_v + (size_t)T_TOK * D_MODEL;

    // ctx = concat(a,v,l) @ fusion_W + fusion_b
    wmma_gemm_fusion<<<(T_TOK / 16) * (D_MODEL / 16) / 8, THREADS, 0, stream>>>(
        xa, xv, xl, fusion_W, fusion_b, ctx);

    run_block(pa, xa, ctx, out_a, u, zx, xcf, xcb, dtf, dtb, ybuf, gbuf, stream);
    run_block(pv, xv, ctx, out_v, u, zx, xcf, xcb, dtf, dtb, ybuf, gbuf, stream);
    run_block(pl, xl, ctx, out_l, u, zx, xcf, xcb, dtf, dtb, ybuf, gbuf, stream);
}
